// VanillaSelfAttention_16578573763414
// MI455X (gfx1250) — matrix operand order repaired and checked by reading, not compiled, not run on hardware
//
#include <hip/hip_runtime.h>

typedef _Float16 h8   __attribute__((ext_vector_type(8)));
typedef _Float16 v16h __attribute__((ext_vector_type(16)));
typedef float    v8f  __attribute__((ext_vector_type(8)));

#define B_      8
#define DIM_    384
#define NH_     12
#define HD_     32
#define HW_     1024
#define QKVO_CH 1536
#define SCALE_F 0.1767766952966369f  // 32^-0.5

// ---------------- WMMA fragment helpers (gfx1250 wave32 layouts) -------------
// A 16x32 f16: lane m=lane%16 is row M; hi=lane/16; per-lane element j:
//   k = hi*16 + j (contiguous, matches frag_b16)
__device__ __forceinline__ v16h frag_a16(const _Float16* row, int hi) {
  h8 lo = *(const h8*)(row + hi * 16);
  h8 hh = *(const h8*)(row + hi * 16 + 8);
  return __builtin_shufflevector(lo, hh, 0,1,2,3,4,5,6,7,8,9,10,11,12,13,14,15);
}
// B 32x16 f16: lane n=lane%16 is column N; element j -> k = hi*16 + j.
// Caller passes p = &col_data[hi*16] with 16 contiguous halfs along K.
__device__ __forceinline__ v16h frag_b16(const _Float16* p) {
  h8 lo = *(const h8*)(p);
  h8 hh = *(const h8*)(p + 8);
  return __builtin_shufflevector(lo, hh, 0,1,2,3,4,5,6,7,8,9,10,11,12,13,14,15);
}
__device__ __forceinline__ v8f wmma_f16(v16h a, v16h b, v8f c) {
  return __builtin_amdgcn_wmma_f32_16x16x32_f16(false, a, false, b, (short)0, c,
                                                false, false);
}

// ---------------- misc converts / transposes ---------------------------------
__global__ __launch_bounds__(256) void k_convert(const float* __restrict__ s,
                                                 _Float16* __restrict__ d, long n) {
  for (long i = (long)blockIdx.x * blockDim.x + threadIdx.x; i < n;
       i += (long)gridDim.x * blockDim.x)
    d[i] = (_Float16)s[i];
}

// v channels of qkvo -> f16 channel-major [B*12][32][1024]
__global__ __launch_bounds__(256) void k_convert_v(const float* __restrict__ qkvo,
                                                   _Float16* __restrict__ vf) {
  long n = (long)B_ * DIM_ * HW_;
  for (long i = (long)blockIdx.x * blockDim.x + threadIdx.x; i < n;
       i += (long)gridDim.x * blockDim.x) {
    long b = i / ((long)DIM_ * HW_);
    long rem = i - b * (long)DIM_ * HW_;
    vf[i] = (_Float16)qkvo[b * (long)QKVO_CH * HW_ + (long)(2 * DIM_) * HW_ + rem];
  }
}

// channel-major f32 [unit][DIM][HW] -> pos-major f16 [unit][HW][DIM]
__global__ __launch_bounds__(1024) void k_transpose_c2p(const float* __restrict__ src,
                                                        _Float16* __restrict__ dst,
                                                        float scale) {
  __shared__ float t[32][33];
  int b = blockIdx.z, c0 = blockIdx.y * 32, p0 = blockIdx.x * 32;
  const float* s = src + ((long)b * DIM_ + c0) * HW_ + p0;
  t[threadIdx.y][threadIdx.x] = s[(long)threadIdx.y * HW_ + threadIdx.x];
  __syncthreads();
  dst[((long)b * HW_ + p0 + threadIdx.y) * DIM_ + c0 + threadIdx.x] =
      (_Float16)(t[threadIdx.x][threadIdx.y] * scale);
}

// head slice of qkvo (ch_base + h*32 .. +32) -> pos-major f16 [bh][1024][32]
__global__ __launch_bounds__(1024) void k_transpose_heads(const float* __restrict__ qkvo,
                                                          _Float16* __restrict__ dst,
                                                          int ch_base, float scale) {
  __shared__ float t[32][33];
  int unit = blockIdx.y;                 // b*12 + h
  int b = unit / NH_, h = unit % NH_;
  int p0 = blockIdx.x * 32;
  const float* s = qkvo + ((long)b * QKVO_CH + ch_base + h * HD_) * HW_ + p0;
  t[threadIdx.y][threadIdx.x] = s[(long)threadIdx.y * HW_ + threadIdx.x];
  __syncthreads();
  dst[((long)unit * HW_ + p0 + threadIdx.y) * HD_ + threadIdx.x] =
      (_Float16)(t[threadIdx.x][threadIdx.y] * scale);
}

// ---------------- f16 WMMA GEMM: out[M][8192] = A[M][K] * Bt^T + bias --------
// A row-major [M][K] f16, Bt pos-major [8192][K] f16, out f32 per-batch
// channel-major: out + b*M*1024 + row*1024 + pos.
// Each wave computes a 16x64 strip: one A fragment reused across 4 B fragments
// (2.5 b128 loads per WMMA instead of 4).
__global__ __launch_bounds__(256) void k_gemm_f16(const _Float16* __restrict__ A,
                                                  const _Float16* __restrict__ Bt,
                                                  const float* __restrict__ bias,
                                                  float* __restrict__ out,
                                                  int M, int K) {
  const int ncg = (B_ * HW_) / 64;  // 128 column groups of 64
  int lane = threadIdx.x & 31, wave = threadIdx.x >> 5;
  int tile = blockIdx.x * 8 + wave;
  int ti = tile / ncg, tg = tile % ncg;
  if (ti * 16 >= M) return;                 // wave-uniform
  int m = lane & 15, hi = lane >> 4;
  int och0 = ti * 16, col0 = tg * 64;
  const _Float16* arow = A + (long)(och0 + m) * K;
  const _Float16* brow0 = Bt + (long)(col0 + m) * K;
  const _Float16* brow1 = Bt + (long)(col0 + 16 + m) * K;
  const _Float16* brow2 = Bt + (long)(col0 + 32 + m) * K;
  const _Float16* brow3 = Bt + (long)(col0 + 48 + m) * K;
  v8f acc0 = {0.f,0.f,0.f,0.f,0.f,0.f,0.f,0.f};
  v8f acc1 = acc0, acc2 = acc0, acc3 = acc0;
  for (int k0 = 0; k0 < K; k0 += 32) {
    v16h a = frag_a16(arow + k0, hi);
    int ko = k0 + hi * 16;
    acc0 = wmma_f16(a, frag_b16(brow0 + ko), acc0);
    acc1 = wmma_f16(a, frag_b16(brow1 + ko), acc1);
    acc2 = wmma_f16(a, frag_b16(brow2 + ko), acc2);
    acc3 = wmma_f16(a, frag_b16(brow3 + ko), acc3);
  }
  int bb = col0 / HW_, pos = col0 % HW_;
  float* obase = out + (long)bb * M * HW_ + (long)och0 * HW_ + pos;
#pragma unroll
  for (int r = 0; r < 8; ++r) {
    int row = r + 8 * hi;
    float bv = bias[och0 + row];
    float* orow = obase + (long)row * HW_;
    orow[m]      = acc0[r] + bv;
    orow[16 + m] = acc1[r] + bv;
    orow[32 + m] = acc2[r] + bv;
    orow[48 + m] = acc3[r] + bv;
  }
}

// ---------------- flash attention (online softmax, WMMA) ---------------------
// qf: [96][1024][32] f16 (pre-scaled), kf: [96][1024][32] f16,
// vf: [96][32][1024] f16 (V^T), resA: [8][384][1024] f32 channel-major.
#define KBLK 128
__global__ __launch_bounds__(256) void k_attn(const _Float16* __restrict__ qf,
                                              const _Float16* __restrict__ kf,
                                              const _Float16* __restrict__ vf,
                                              float* __restrict__ resA) {
  __shared__ _Float16 Kls[KBLK][48];    // rows of K block, padded (16B mult)
  __shared__ _Float16 Vls[32][136];     // V^T block rows (d), padded
  __shared__ _Float16 Pls[8][16][48];   // per-wave P tile staging
  __shared__ float    Ols[8][32][17];   // per-wave output transpose staging

  int bh = blockIdx.x;                  // b*12 + h
  int qb = blockIdx.y;                  // 0..7 (128-query slab)
  int tid = threadIdx.x, lane = tid & 31, wave = tid >> 5;
  int m = lane & 15, hi = lane >> 4;

  const _Float16* qh = qf + (long)bh * HW_ * HD_;
  const _Float16* kh = kf + (long)bh * HW_ * HD_;
  const _Float16* vh = vf + (long)bh * HD_ * HW_;
  int q0 = qb * 128 + wave * 16;

  v16h aq = frag_a16(qh + (long)(q0 + m) * HD_, hi);  // Q tile (constant)
  v8f o0 = {0.f,0.f,0.f,0.f,0.f,0.f,0.f,0.f};
  v8f o1 = {0.f,0.f,0.f,0.f,0.f,0.f,0.f,0.f};
  float mrow[8], lrow[8];
#pragma unroll
  for (int r = 0; r < 8; ++r) { mrow[r] = -1e30f; lrow[r] = 0.f; }

  for (int kb = 0; kb < HW_; kb += KBLK) {
    __syncthreads();
    for (int c = tid; c < 512; c += 256) {          // K block: 128x32 halfs
      int row = c >> 2, part = c & 3;
      *(h8*)&Kls[row][part * 8] = *(const h8*)(kh + (long)(kb + row) * HD_ + part * 8);
    }
    for (int c = tid; c < 512; c += 256) {          // V^T block: 32x128 halfs
      int row = c >> 4, part = c & 15;
      *(h8*)&Vls[row][part * 8] = *(const h8*)(vh + (long)row * HW_ + kb + part * 8);
    }
    __syncthreads();

    for (int ks = 0; ks < KBLK; ks += 32) {
      v16h b0 = frag_b16(&Kls[ks + m][hi * 16]);
      v16h b1 = frag_b16(&Kls[ks + 16 + m][hi * 16]);
      v8f z = {0.f,0.f,0.f,0.f,0.f,0.f,0.f,0.f};
      v8f s0 = wmma_f16(aq, b0, z);
      v8f s1 = wmma_f16(aq, b1, z);

      float alpha[8];
#pragma unroll
      for (int r = 0; r < 8; ++r) {                 // row max over 16 lanes
        float v = fmaxf(s0[r], s1[r]);
        v = fmaxf(v, __shfl_xor(v, 1, 32));
        v = fmaxf(v, __shfl_xor(v, 2, 32));
        v = fmaxf(v, __shfl_xor(v, 4, 32));
        v = fmaxf(v, __shfl_xor(v, 8, 32));
        float mn = fmaxf(mrow[r], v);
        alpha[r] = __expf(mrow[r] - mn);
        mrow[r] = mn;
      }
#pragma unroll
      for (int r = 0; r < 8; ++r) {
        float p0 = __expf(s0[r] - mrow[r]);
        float p1 = __expf(s1[r] - mrow[r]);
        Pls[wave][r + 8 * hi][m]      = (_Float16)p0;
        Pls[wave][r + 8 * hi][16 + m] = (_Float16)p1;
        float sum = p0 + p1;
        sum += __shfl_xor(sum, 1, 32);
        sum += __shfl_xor(sum, 2, 32);
        sum += __shfl_xor(sum, 4, 32);
        sum += __shfl_xor(sum, 8, 32);
        lrow[r] = lrow[r] * alpha[r] + sum;
        o0[r] *= alpha[r];
        o1[r] *= alpha[r];
      }
      v16h ap  = frag_a16(&Pls[wave][m][0], hi);    // P as A operand (16x32)
      v16h bv0 = frag_b16(&Vls[m][ks + hi * 16]);   // V^T rows d=0..15
      v16h bv1 = frag_b16(&Vls[16 + m][ks + hi * 16]);
      o0 = wmma_f16(ap, bv0, o0);
      o1 = wmma_f16(ap, bv1, o1);
    }
  }

#pragma unroll
  for (int r = 0; r < 8; ++r) {
    float inv = 1.0f / lrow[r];
    Ols[wave][m][r + 8 * hi]      = o0[r] * inv;    // transpose to [d][q]
    Ols[wave][16 + m][r + 8 * hi] = o1[r] * inv;
  }
  int b = bh / NH_, h = bh % NH_;
  float* rbase = resA + ((long)b * DIM_ + h * HD_) * HW_ + q0 + (long)lane * HW_;
#pragma unroll
  for (int i4 = 0; i4 < 4; ++i4) {
    float4 t = make_float4(Ols[wave][lane][4 * i4 + 0], Ols[wave][lane][4 * i4 + 1],
                           Ols[wave][lane][4 * i4 + 2], Ols[wave][lane][4 * i4 + 3]);
    *(float4*)(rbase + 4 * i4) = t;
  }
}

// ---------------- lepe depthwise 5x5 + residual + *o -------------------------
__global__ __launch_bounds__(256) void k_lepe_combine(const float* __restrict__ qkvo,
                                                      const float* __restrict__ lw,
                                                      const float* __restrict__ lb,
                                                      const float* __restrict__ resA,
                                                      float* __restrict__ u) {
  int b = blockIdx.x / DIM_, c = blockIdx.x % DIM_;
  const float* vch = qkvo + ((long)b * QKVO_CH + 2 * DIM_ + c) * HW_;
  const float* och = qkvo + ((long)b * QKVO_CH + 3 * DIM_ + c) * HW_;
  const float* rch = resA + ((long)b * DIM_ + c) * HW_;
  float* uch = u + ((long)b * DIM_ + c) * HW_;
  __shared__ float tile[36][36];
  __shared__ float w[25];
  if (threadIdx.x < 25) w[threadIdx.x] = lw[c * 25 + threadIdx.x];
  for (int i = threadIdx.x; i < 36 * 36; i += 256) {
    int y = i / 36 - 2, x = i % 36 - 2;
    float v = 0.f;
    if (y >= 0 && y < 32 && x >= 0 && x < 32) v = vch[y * 32 + x];
    tile[i / 36][i % 36] = v;
  }
  __syncthreads();
  float bias = lb[c];
  for (int p = threadIdx.x; p < HW_; p += 256) {
    int y = p >> 5, x = p & 31;
    float acc = bias;
#pragma unroll
    for (int ky = 0; ky < 5; ++ky)
#pragma unroll
      for (int kx = 0; kx < 5; ++kx)
        acc += tile[y + ky][x + kx] * w[ky * 5 + kx];
    uch[p] = (rch[p] + acc) * och[p];
  }
}

// ---------------- host ------------------------------------------------------
extern "C" void kernel_launch(void* const* d_in, const int* in_sizes, int n_in,
                              void* d_out, int out_size, void* d_ws, size_t ws_size,
                              hipStream_t stream) {
  const float* x      = (const float*)d_in[0];
  const float* qkvo_w = (const float*)d_in[1];
  const float* qkvo_b = (const float*)d_in[2];
  const float* lepe_w = (const float*)d_in[3];
  const float* lepe_b = (const float*)d_in[4];
  const float* proj_w = (const float*)d_in[5];
  const float* proj_b = (const float*)d_in[6];
  float* out = (float*)d_out;

  char* p = (char*)d_ws;
  auto carve = [&](size_t bytes) -> char* {
    char* r = p;
    p += (bytes + 255) & ~(size_t)255;
    return r;
  };
  _Float16* wq   = (_Float16*)carve((size_t)QKVO_CH * DIM_ * 2);
  _Float16* wp   = (_Float16*)carve((size_t)DIM_ * DIM_ * 2);
  _Float16* xt   = (_Float16*)carve((size_t)B_ * HW_ * DIM_ * 2);
  float*    qkvo = (float*)   carve((size_t)B_ * QKVO_CH * HW_ * 4);
  _Float16* qfh  = (_Float16*)carve((size_t)B_ * DIM_ * HW_ * 2);
  _Float16* kfh  = (_Float16*)carve((size_t)B_ * DIM_ * HW_ * 2);
  _Float16* vfh  = (_Float16*)carve((size_t)B_ * DIM_ * HW_ * 2);
  float*    resA = (float*)   carve((size_t)B_ * DIM_ * HW_ * 4);
  float*    u    = (float*)   carve((size_t)B_ * DIM_ * HW_ * 4);
  _Float16* tf   = (_Float16*)carve((size_t)B_ * HW_ * DIM_ * 2);

  dim3 tb(32, 32);

  k_convert<<<2304, 256, 0, stream>>>(qkvo_w, wq, (long)QKVO_CH * DIM_);
  k_convert<<<576, 256, 0, stream>>>(proj_w, wp, (long)DIM_ * DIM_);
  k_transpose_c2p<<<dim3(32, DIM_ / 32, B_), tb, 0, stream>>>(x, xt, 1.0f);

  // 16x64 strips: (1536/16)*(8192/64) = 96*128 = 12288 wave-tiles / 8 waves
  k_gemm_f16<<<1536, 256, 0, stream>>>(wq, xt, qkvo_b, qkvo, QKVO_CH, DIM_);

  k_transpose_heads<<<dim3(32, B_ * NH_), tb, 0, stream>>>(qkvo, qfh, 0, SCALE_F);
  k_transpose_heads<<<dim3(32, B_ * NH_), tb, 0, stream>>>(qkvo, kfh, DIM_, 1.0f);
  k_convert_v<<<12288, 256, 0, stream>>>(qkvo, vfh);

  k_attn<<<dim3(B_ * NH_, 8), 256, 0, stream>>>(qfh, kfh, vfh, resA);

  k_lepe_combine<<<B_ * DIM_, 256, 0, stream>>>(qkvo, lepe_w, lepe_b, resA, u);
  k_transpose_c2p<<<dim3(32, DIM_ / 32, B_), tb, 0, stream>>>(u, tf, 1.0f);

  // (384/16)*(8192/64) = 24*128 = 3072 wave-tiles / 8 waves
  k_gemm_f16<<<384, 256, 0, stream>>>(wp, tf, proj_b, out, DIM_, DIM_);
}
